// ScaledDotProductAttention_8564164789040
// MI455X (gfx1250) — matrix operand order repaired and checked by reading, not compiled, not run on hardware
//
#include <hip/hip_runtime.h>
#include <hip/hip_bf16.h>

typedef __attribute__((ext_vector_type(16))) _Float16 v16h;
typedef __attribute__((ext_vector_type(8)))  float    v8f;

constexpr int L    = 2048;   // sequence length
constexpr int D    = 128;    // head dim
constexpr int SROW = 2052;   // padded LDS row stride (floats): 8*SROW % 64 == 32 -> no
                             // bank clash between the two half-wave groups on C writes

__launch_bounds__(256)
__global__ void sdpa_wmma_kernel(const float* __restrict__ Qm,
                                 const float* __restrict__ Km,
                                 const float* __restrict__ Vm,
                                 float* __restrict__ ctx,
                                 float* __restrict__ attn)
{
    extern __shared__ float sS[];            // [16][SROW] fp32 scores -> probabilities

    const int b     = blockIdx.y;
    const int qbase = blockIdx.x * 16;
    const int tid   = threadIdx.x;
    const int wave  = tid >> 5;              // 8 waves of 32 (wave32)
    const int lane  = tid & 31;
    const int m     = lane & 15;             // A-row / C-column index
    const int g     = lane >> 4;             // half-wave group

    const size_t bL = (size_t)b * L;

    // ---------------- Phase 1: per-wave Q A-fragments (16x32 f16, 4 chunks = D) ----
    // A layout (ISA 7.12.2, 16-bit A 16x32): lane(0..15)->M, VGPR v holds K pair
    //   base = (v<4 ? 2v : 16+2(v-4)) + 8*g  (g = lane>=16 group)
    const float qscale = 0.088388347648318447f;     // 1/sqrt(128)
    const float* qrow = Qm + (bL + qbase + m) * D;
    v16h qa[4];
#pragma unroll
    for (int c = 0; c < 4; ++c) {
#pragma unroll
        for (int vv = 0; vv < 8; ++vv) {
            const int kk = c * 32 + ((vv & 3) * 2) + ((vv >> 2) * 16) + g * 8;
            const float2 f = *(const float2*)(qrow + kk);
            qa[c][2 * vv]     = (_Float16)(f.x * qscale);
            qa[c][2 * vv + 1] = (_Float16)(f.y * qscale);
        }
    }

    // ---------------- Phase 2: S = (Q/sqrt(d)) K^T, waves split the 128 key tiles --
    // B layout: lane -> column N=m (key), element i -> K (dim within 32-chunk).
    // Two independent accumulation chains (c={0,2} and c={1,3}) so WMMAs of one
    // chain overlap fragment-conversion VALU of the other (avoids WAR hazard NOPs).
    for (int kt = wave; kt < L / 16; kt += 8) {
        const float* kbase = Km + (bL + (size_t)kt * 16) * D;
        v16h bf[4];
#pragma unroll
        for (int c = 0; c < 4; ++c)
#pragma unroll
            for (int i = 0; i < 16; ++i)     // lane -> key N=m, element i -> dim K (table order)
                bf[c][i] = (_Float16)kbase[(size_t)m * D + c * 32 + (i & 7) + ((i >> 3) * 16) + g * 8];

        v8f acc0 = {}, acc1 = {};
        acc0 = __builtin_amdgcn_wmma_f32_16x16x32_f16(false, qa[0], false, bf[0],
                                                      (short)0, acc0, false, false);
        acc1 = __builtin_amdgcn_wmma_f32_16x16x32_f16(false, qa[1], false, bf[1],
                                                      (short)0, acc1, false, false);
        acc0 = __builtin_amdgcn_wmma_f32_16x16x32_f16(false, qa[2], false, bf[2],
                                                      (short)0, acc0, false, false);
        acc1 = __builtin_amdgcn_wmma_f32_16x16x32_f16(false, qa[3], false, bf[3],
                                                      (short)0, acc1, false, false);
        const v8f acc = acc0 + acc1;

        if (kt + 8 < L / 16)                 // global_prefetch_b8 for next key tile
            __builtin_prefetch(Km + (bL + (size_t)(kt + 8) * 16) * D + lane * 4, 0, 1);
        // C layout: lane -> N (=m), VGPR r -> M = r + 8*g
#pragma unroll
        for (int r = 0; r < 8; ++r)
            sS[(r + 8 * g) * SROW + kt * 16 + m] = acc[r];
    }
    __syncthreads();

    // ---------------- Phase 3: row softmax + stream attention out (NT, 512 MB) -----
#pragma unroll
    for (int t = 0; t < 2; ++t) {
        const int rr = 2 * wave + t;
        float* srow = sS + rr * SROW;

        float mx = -3.402823466e38f;
        for (int j = lane; j < L; j += 32) mx = fmaxf(mx, srow[j]);
#pragma unroll
        for (int off = 16; off; off >>= 1) mx = fmaxf(mx, __shfl_xor(mx, off, 32));

        float sum = 0.f;
        for (int j = lane; j < L; j += 32) {
            const float e = __expf(srow[j] - mx);
            srow[j] = e;
            sum += e;
        }
#pragma unroll
        for (int off = 16; off; off >>= 1) sum += __shfl_xor(sum, off, 32);
        const float inv = __frcp_rn(sum);

        float* arow = attn + (bL + qbase + rr) * L;
        for (int j = lane; j < L; j += 32) {
            const float p = srow[j] * inv;
            srow[j] = p;                               // keep P for phase 4
            __builtin_nontemporal_store(p, arow + j);  // TH=NT: don't pollute L2
        }
    }
    __syncthreads();

    // ---------------- Phase 4: context = P V ; wave w owns dims [16w, 16w+16) ------
    // Unrolled x2 with two accumulators: breaks the D->C serial chain and gives the
    // scheduler independent fragment registers (no WMMA<->VALU WAR hazard stalls).
    {
        const int dbase = wave * 16;
        v8f acc[2] = {{}, {}};
        for (int kt = 0; kt < L / 32; kt += 2) {
#pragma unroll
            for (int s = 0; s < 2; ++s) {
                const int kb = (kt + s) * 32;
                v16h pa, vb;
#pragma unroll
                for (int vv = 0; vv < 8; ++vv) {       // A from LDS probabilities
                    const int kk = kb + ((vv & 3) * 2) + ((vv >> 2) * 16) + g * 8;
                    pa[2 * vv]     = (_Float16)sS[m * SROW + kk];
                    pa[2 * vv + 1] = (_Float16)sS[m * SROW + kk + 1];
                }
                const float* vrow = Vm + (bL + (size_t)kb) * D + dbase + m;
#pragma unroll
                for (int i = 0; i < 16; ++i)           // B: lane->dim m, elem i->key (table order)
                    vb[i] = (_Float16)vrow[(size_t)((i & 7) + ((i >> 3) * 16) + g * 8) * D];
                acc[s] = __builtin_amdgcn_wmma_f32_16x16x32_f16(false, pa, false, vb,
                                                                (short)0, acc[s],
                                                                false, false);
            }
        }
        const v8f accf = acc[0] + acc[1];
        float* crow = ctx + (bL + qbase) * D + dbase;
#pragma unroll
        for (int r = 0; r < 8; ++r)
            __builtin_nontemporal_store(accf[r], crow + (size_t)(r + 8 * g) * D + m);
    }
}

extern "C" void kernel_launch(void* const* d_in, const int* in_sizes, int n_in,
                              void* d_out, int out_size, void* d_ws, size_t ws_size,
                              hipStream_t stream)
{
    const float* q = (const float*)d_in[0];
    const float* k = (const float*)d_in[1];
    const float* v = (const float*)d_in[2];
    const int B = in_sizes[0] / (L * D);               // 32 for the reference shapes

    float* ctx  = (float*)d_out;                       // [B, L, D]
    float* attn = ctx + (size_t)B * L * D;             // [B, L, L]

    dim3 grid(L / 16, B);                              // batch-major => L2 reuse of K/V
    const size_t smem = (size_t)16 * SROW * sizeof(float);   // ~128 KB dynamic LDS
    sdpa_wmma_kernel<<<grid, dim3(256), smem, stream>>>(q, k, v, ctx, attn);
}
